// TypesMLP_46720654246527
// MI455X (gfx1250) — matrix operand order repaired and checked by reading, not compiled, not run on hardware
//
#include <hip/hip_runtime.h>
#include <hip/hip_bf16.h>

#define NATOMS 65536
#define DIN    256
#define HID    512
#define NSPEC  4
#define BM     32   // atoms per block

typedef _Float16 h8   __attribute__((ext_vector_type(8)));
typedef _Float16 v16h __attribute__((ext_vector_type(16)));
typedef float    v8f  __attribute__((ext_vector_type(8)));

// ---------------- prep kernels ----------------

__global__ void zero_cnt_kernel(int* __restrict__ cnt) {
    if (threadIdx.x < NSPEC) cnt[threadIdx.x] = 0;
}

__global__ void bin_atoms_kernel(const int* __restrict__ types,
                                 int* __restrict__ cnt,
                                 int* __restrict__ idxbuf) {
    int i = blockIdx.x * blockDim.x + threadIdx.x;
    if (i < NATOMS) {
        int t = types[i];
        int slot = atomicAdd(&cnt[t], 1);
        idxbuf[t * NATOMS + slot] = i;
    }
}

// W1 [S][DIN][HID] f32  ->  W1t [S][HID][DIN] f16 (transposed, K contiguous)
__global__ void conv_w1_kernel(const float* __restrict__ W1, _Float16* __restrict__ W1t) {
    int i = blockIdx.x * blockDim.x + threadIdx.x;
    if (i >= NSPEC * HID * DIN) return;
    int sp  = i / (HID * DIN);
    int rem = i % (HID * DIN);
    int n   = rem / DIN;
    int k   = rem % DIN;
    W1t[i] = (_Float16)W1[((size_t)(sp * DIN + k)) * HID + n];
}

// W2 [S][HID][HID] f32  ->  W2t [S][HID][HID] f16 (transposed)
__global__ void conv_w2_kernel(const float* __restrict__ W2, _Float16* __restrict__ W2t) {
    int i = blockIdx.x * blockDim.x + threadIdx.x;
    if (i >= NSPEC * HID * HID) return;
    int sp  = i / (HID * HID);
    int rem = i % (HID * HID);
    int n   = rem / HID;
    int k   = rem % HID;
    W2t[i] = (_Float16)W2[((size_t)(sp * HID + k)) * HID + n];
}

// ---------------- main kernel ----------------

__device__ __forceinline__ v16h make_frag(const _Float16* lo, const _Float16* hi) {
    h8 a = *(const h8*)lo;
    h8 b = *(const h8*)hi;
    return __builtin_shufflevector(a, b, 0,1,2,3,4,5,6,7,8,9,10,11,12,13,14,15);
}

__global__ __launch_bounds__(256) void types_mlp_kernel(
    const float*    __restrict__ features,
    const float*    __restrict__ b1,
    const float*    __restrict__ b2,
    const float*    __restrict__ W3,
    const float*    __restrict__ b3,
    const _Float16* __restrict__ W1t,
    const _Float16* __restrict__ W2t,
    const int*      __restrict__ cnt,
    const int*      __restrict__ idxbuf,
    float*          __restrict__ out)
{
    // bufA: X tile (BM x DIN f16, first 16KB) during layer1, then H2 (BM x HID) after
    // bufB: H1 (BM x HID)
    __shared__ __align__(16) _Float16 bufA[BM * HID];
    __shared__ __align__(16) _Float16 bufB[BM * HID];

    const int sp   = blockIdx.y;
    const int row0 = blockIdx.x * BM;
    const int csp  = cnt[sp];
    if (row0 >= csp) return;                 // uniform early exit

    const int tid = threadIdx.x;

    // ---- stage X: gather 32 atom rows (f32 -> f16 into bufA, stride DIN) ----
    {
        int r  = tid >> 3;                   // row 0..31 (8 threads per row)
        int cb = (tid & 7) * 32;             // 32 columns per thread
        int slot = row0 + r;
        int g = (slot < csp) ? idxbuf[sp * NATOMS + slot] : -1;
        if (g >= 0) {
            const float4* src = (const float4*)(features + (size_t)g * DIN + cb);
            #pragma unroll
            for (int j = 0; j < 8; ++j) {
                float4 v = src[j];
                int c = cb + j * 4;
                bufA[r * DIN + c + 0] = (_Float16)v.x;
                bufA[r * DIN + c + 1] = (_Float16)v.y;
                bufA[r * DIN + c + 2] = (_Float16)v.z;
                bufA[r * DIN + c + 3] = (_Float16)v.w;
            }
        } else {
            #pragma unroll
            for (int j = 0; j < 32; ++j) bufA[r * DIN + cb + j] = (_Float16)0.0f;
        }
    }
    __syncthreads();

    const int lane = tid & 31;
    const int wave = tid >> 5;               // 8 waves
    const int mt   = wave & 1;               // M tile (rows mt*16..+15)
    const int ng   = wave >> 1;              // N group of 128 columns
    const int lm   = lane & 15;
    const int lhi  = lane >> 4;              // 0/1
    const int m    = mt * 16 + lm;           // this lane's A row

    v8f acc[8];
    const v8f vzero = {0.f,0.f,0.f,0.f,0.f,0.f,0.f,0.f};

    // ================= layer 1: [32 x 256] x [256 x 512] =================
    #pragma unroll
    for (int i = 0; i < 8; ++i) acc[i] = vzero;

    for (int kc = 0; kc < DIN; kc += 32) {
        int ak = kc + lhi * 8;
        v16h a = make_frag(&bufA[m * DIN + ak], &bufA[m * DIN + ak + 16]);
        #pragma unroll
        for (int nt = 0; nt < 8; ++nt) {
            int n = ng * 128 + nt * 16 + lm;
            const _Float16* wp = W1t + ((size_t)(sp * HID + n)) * DIN + kc + lhi * 8;
            v16h b = make_frag(wp, wp + 16);
            acc[nt] = __builtin_amdgcn_wmma_f32_16x16x32_f16(
                false, a, false, b, (short)0, acc[nt], false, false);
        }
    }
    // bias + tanh -> bufB (H1, stride HID)
    #pragma unroll
    for (int nt = 0; nt < 8; ++nt) {
        int n = ng * 128 + nt * 16 + lm;
        float bias = b1[sp * HID + n];
        #pragma unroll
        for (int r = 0; r < 8; ++r) {
            int mm = mt * 16 + lhi * 8 + r;
            bufB[mm * HID + n] = (_Float16)tanhf(acc[nt][r] + bias);
        }
    }
    __syncthreads();   // H1 complete; X (bufA) no longer needed

    // ================= layer 2: [32 x 512] x [512 x 512] =================
    #pragma unroll
    for (int i = 0; i < 8; ++i) acc[i] = vzero;

    for (int kc = 0; kc < HID; kc += 32) {
        int ak = kc + lhi * 8;
        v16h a = make_frag(&bufB[m * HID + ak], &bufB[m * HID + ak + 16]);
        #pragma unroll
        for (int nt = 0; nt < 8; ++nt) {
            int n = ng * 128 + nt * 16 + lm;
            const _Float16* wp = W2t + ((size_t)(sp * HID + n)) * HID + kc + lhi * 8;
            v16h b = make_frag(wp, wp + 16);
            acc[nt] = __builtin_amdgcn_wmma_f32_16x16x32_f16(
                false, a, false, b, (short)0, acc[nt], false, false);
        }
    }
    // bias + tanh -> bufA (H2, stride HID)
    #pragma unroll
    for (int nt = 0; nt < 8; ++nt) {
        int n = ng * 128 + nt * 16 + lm;
        float bias = b2[sp * HID + n];
        #pragma unroll
        for (int r = 0; r < 8; ++r) {
            int mm = mt * 16 + lhi * 8 + r;
            bufA[mm * HID + n] = (_Float16)tanhf(acc[nt][r] + bias);
        }
    }
    __syncthreads();   // H2 complete

    // ================= layer 3: [32 x 512] . [512] -> scatter =================
    #pragma unroll
    for (int rr = 0; rr < 4; ++rr) {
        int mm = wave * 4 + rr;              // 8 waves x 4 rows = 32 rows
        float p = 0.0f;
        #pragma unroll
        for (int k = lane; k < HID; k += 32)
            p += (float)bufA[mm * HID + k] * W3[sp * HID + k];
        #pragma unroll
        for (int off = 16; off > 0; off >>= 1)
            p += __shfl_xor(p, off, 32);
        if (lane == 0) {
            int slot = row0 + mm;
            if (slot < csp) {
                int g = idxbuf[sp * NATOMS + slot];
                out[g] = p + b3[sp];
            }
        }
    }
}

// ---------------- host launcher ----------------

extern "C" void kernel_launch(void* const* d_in, const int* in_sizes, int n_in,
                              void* d_out, int out_size, void* d_ws, size_t ws_size,
                              hipStream_t stream) {
    const float* features = (const float*)d_in[0];
    const int*   types    = (const int*)  d_in[1];
    const float* W1       = (const float*)d_in[2];
    const float* b1       = (const float*)d_in[3];
    const float* W2       = (const float*)d_in[4];
    const float* b2       = (const float*)d_in[5];
    const float* W3       = (const float*)d_in[6];
    const float* b3       = (const float*)d_in[7];
    float*       out      = (float*)d_out;

    char* ws = (char*)d_ws;
    int*       cnt    = (int*)ws;                                   // 256 B slot
    int*       idxbuf = (int*)(ws + 256);                           // S*N ints = 1 MB
    _Float16*  W1t    = (_Float16*)(ws + 256 + sizeof(int) * (size_t)NSPEC * NATOMS);
    _Float16*  W2t    = W1t + (size_t)NSPEC * HID * DIN;            // +1 MB, then 2 MB

    zero_cnt_kernel<<<1, 32, 0, stream>>>(cnt);
    bin_atoms_kernel<<<NATOMS / 256, 256, 0, stream>>>(types, cnt, idxbuf);
    conv_w1_kernel<<<(NSPEC * HID * DIN + 255) / 256, 256, 0, stream>>>(W1, W1t);
    conv_w2_kernel<<<(NSPEC * HID * HID + 255) / 256, 256, 0, stream>>>(W2, W2t);

    dim3 grid(NATOMS / BM, NSPEC);
    types_mlp_kernel<<<grid, 256, 0, stream>>>(features, b1, b2, W3, b3,
                                               W1t, W2t, cnt, idxbuf, out);
}
